// Qwen2_5_VLVisionModel_16097537425623
// MI455X (gfx1250) — matrix operand order repaired and checked by reading, not compiled, not run on hardware
//
#include <hip/hip_runtime.h>

typedef __attribute__((ext_vector_type(16))) _Float16 v16h;
typedef __attribute__((ext_vector_type(4)))  _Float16 h4;
typedef __attribute__((ext_vector_type(8)))  float    v8f;
typedef unsigned int u32;
typedef __attribute__((ext_vector_type(4)))  u32      u4;

#define WMMA_F16(A, B, C) \
  __builtin_amdgcn_wmma_f32_16x16x32_f16(false, (A), false, (B), (short)0, (C), false, false)

// ---------------------------------------------------------------------------
// Model dimensions
// ---------------------------------------------------------------------------
#define NTOK   1600
#define DMODEL 1280
#define NHEADS 16
#define HDIM   80
#define DEPTH  4
#define INTER  3420
#define OUTD   3584
#define PATCH_IN 1176
#define MROWS  400          // merged rows
#define MDIM   5120         // D * MERGE^2

// Convert float4 -> 4 halves, scaling by s (also used as a 0/1 validity mask
// so out-of-bounds staging loads are zeroed arithmetically, keeping the
// global_load_b128 unconditional and batchable).
__device__ __forceinline__ h4 cvt4(float4 v, float s) {
  h4 r;
  r[0] = (_Float16)(v.x * s);
  r[1] = (_Float16)(v.y * s);
  r[2] = (_Float16)(v.z * s);
  r[3] = (_Float16)(v.w * s);
  return r;
}

// Swizzle for A-type (M x K) tiles: within each 32-wide K chunk, permute the
// 8-half blocks [0,1,2,3] -> stored order [0,2,1,3] so that a lane's fragment
// halves (K blocks {0-7,16-23} for laneHi=0, {8-15,24-31} for laneHi=1) are
// 32 contiguous bytes at column laneHi*16.
__device__ __forceinline__ int swz(int col) {
  const int blk  = (col >> 3) & 3;
  const int perm = ((blk & 1) << 1) | (blk >> 1);   // 0,2,1,3
  return (col & ~31) | (perm << 3) | (col & 7);
}

// Load one 16x16x32 fragment (16 halves) from 32 contiguous LDS bytes.
__device__ __forceinline__ v16h load_frag32(const _Float16* p) {
  v16h f;
  ((u4*)&f)[0] = ((const u4*)p)[0];
  ((u4*)&f)[1] = ((const u4*)p)[1];
  return f;
}

// ---------------------------------------------------------------------------
// Generic f16-WMMA GEMM:  C[M,N] = act( A[M,K] @ W[N,K]^T + bias ) + residual
// Block = 128 threads (4 waves), tile 64(M)x128(N), K-chunks of 32.
// Each wave computes 32x64 = 2x4 WMMA fragments (8 wmma per chunk).
// Staging: phase-1 batched unconditional clamped float4 loads, phase-2
// mask-scaled f16 convert + ds_store_b64. (All K are multiples of 4.)
// ---------------------------------------------------------------------------
__global__ __launch_bounds__(128) void gemm_f16_kernel(
    const float* __restrict__ A, const float* __restrict__ W,
    const float* __restrict__ bias, const float* __restrict__ residual,
    float* __restrict__ C, int M, int N, int K, int act)
{
  __shared__ _Float16 As[64][32];    // stored swizzled (A-type)
  __shared__ _Float16 Ws[128][32];   // stored swizzled (same as As)

  const int tid    = threadIdx.x;
  const int lane   = tid & 31;
  const int wave   = tid >> 5;
  const int laneLo = lane & 15;
  const int laneHi = lane >> 4;
  const int bm = blockIdx.x * 64;
  const int bn = blockIdx.y * 128;
  const int wm = (wave >> 1) * 32;   // wave tile inside block (M): 0 or 32
  const int wn = (wave & 1) * 64;    // wave tile inside block (N): 0 or 64

  v8f acc[2][4] = {};

  for (int k0 = 0; k0 < K; k0 += 32) {
    __syncthreads();
    // ---- phase 1: batched loads (unconditional, clamped addresses) ----
    float4 va[4], vw[8];
#pragma unroll
    for (int it = 0; it < 4; ++it) {
      const int i = tid + it * 128;
      const int r = i >> 3, c4 = i & 7;
      const int gm = bm + r, gk = k0 + c4 * 4;
      const int cm = (gm < M) ? gm : 0;
      const int ck = (gk + 4 <= K) ? gk : 0;
      va[it] = *(const float4*)(A + (size_t)cm * K + ck);
    }
#pragma unroll
    for (int it = 0; it < 8; ++it) {
      const int i = tid + it * 128;
      const int r = i >> 3, c4 = i & 7;
      const int gn = bn + r, gk = k0 + c4 * 4;
      const int cn = (gn < N) ? gn : 0;
      const int ck = (gk + 4 <= K) ? gk : 0;
      vw[it] = *(const float4*)(W + (size_t)cn * K + ck);
    }
    // ---- phase 2: mask-scaled convert + LDS stores ----
#pragma unroll
    for (int it = 0; it < 4; ++it) {
      const int i = tid + it * 128;
      const int r = i >> 3, c4 = i & 7;
      const int gm = bm + r, gk = k0 + c4 * 4;
      const float msk = ((gm < M) && (gk + 4 <= K)) ? 1.0f : 0.0f;
      *(h4*)&As[r][swz(c4 * 4)] = cvt4(va[it], msk);
    }
#pragma unroll
    for (int it = 0; it < 8; ++it) {
      const int i = tid + it * 128;
      const int r = i >> 3, c4 = i & 7;
      const int gn = bn + r, gk = k0 + c4 * 4;
      const float msk = ((gn < N) && (gk + 4 <= K)) ? 1.0f : 0.0f;
      *(h4*)&Ws[r][swz(c4 * 4)] = cvt4(vw[it], msk);
    }
    __syncthreads();

    // ---- fragments: 32 contiguous bytes each ----
    v16h afrag[2], bfrag[4];
#pragma unroll
    for (int s = 0; s < 2; ++s)
      afrag[s] = load_frag32(&As[wm + s * 16 + laneLo][laneHi * 16]);
#pragma unroll
    for (int s = 0; s < 4; ++s)
      bfrag[s] = load_frag32(&Ws[wn + s * 16 + laneLo][laneHi * 16]);

#pragma unroll
    for (int i = 0; i < 2; ++i)
#pragma unroll
      for (int jn = 0; jn < 4; ++jn)
        acc[i][jn] = WMMA_F16(afrag[i], bfrag[jn], acc[i][jn]);
  }

  // epilogue: C layout -> lane<16: (M=j, N=lane); lane>=16: (M=j+8, N=lane-16)
#pragma unroll
  for (int i = 0; i < 2; ++i) {
#pragma unroll
    for (int jn = 0; jn < 4; ++jn) {
      const int gn = bn + wn + jn * 16 + laneLo;
      const float bv = (bias && gn < N) ? bias[gn] : 0.0f;
#pragma unroll
      for (int j = 0; j < 8; ++j) {
        const int gm = bm + wm + i * 16 + j + laneHi * 8;
        if (gm < M && gn < N) {
          float v = acc[i][jn][j] + bv;
          if (act == 1)  // exact GELU
            v = 0.5f * v * (1.0f + erff(v * 0.70710678118654752f));
          if (residual) v += residual[(size_t)gm * N + gn];
          C[(size_t)gm * N + gn] = v;
        }
      }
    }
  }
}

// ---------------------------------------------------------------------------
// RMSNorm: out[row] = x[row] * rsqrt(mean(x^2)+eps) * w   (D % 4 == 0)
// ---------------------------------------------------------------------------
__global__ __launch_bounds__(256) void rmsnorm_kernel(
    const float* __restrict__ x, const float* __restrict__ w,
    float* __restrict__ out, int D)
{
  const int row = blockIdx.x;
  const float* xr = x + (size_t)row * D;
  float ss = 0.0f;
  for (int i4 = threadIdx.x; i4 * 4 < D; i4 += 256) {
    const float4 v = *(const float4*)(xr + i4 * 4);
    ss += v.x * v.x + v.y * v.y + v.z * v.z + v.w * v.w;
  }
#pragma unroll
  for (int off = 16; off > 0; off >>= 1) ss += __shfl_xor(ss, off, 32);
  __shared__ float red[8];
  if ((threadIdx.x & 31) == 0) red[threadIdx.x >> 5] = ss;
  __syncthreads();
  if (threadIdx.x < 8) {
    float v = red[threadIdx.x];
#pragma unroll
    for (int off = 4; off > 0; off >>= 1) v += __shfl_xor(v, off, 8);
    if (threadIdx.x == 0) red[0] = v;
  }
  __syncthreads();
  const float inv = rsqrtf(red[0] / (float)D + 1e-6f);
  for (int i4 = threadIdx.x; i4 * 4 < D; i4 += 256) {
    const float4 v = *(const float4*)(xr + i4 * 4);
    const float4 g = *(const float4*)(w + i4 * 4);
    float4 o;
    o.x = v.x * inv * g.x; o.y = v.y * inv * g.y;
    o.z = v.z * inv * g.z; o.w = v.w * inv * g.w;
    *(float4*)(out + (size_t)row * D + i4 * 4) = o;
  }
}

// ---------------------------------------------------------------------------
// RoPE applied in-place to q,k inside qkv[N][3*D]. One thread per (n,qk,h,d<40)
// pair so both halves are read before either is written.
// ---------------------------------------------------------------------------
__global__ void rope_kernel(float* __restrict__ qkv,
                            const float* __restrict__ cosp,
                            const float* __restrict__ sinp)
{
  int idx = blockIdx.x * blockDim.x + threadIdx.x;   // N*2*NH*40
  if (idx >= NTOK * 2 * NHEADS * 40) return;
  const int d     = idx % 40;
  const int h     = (idx / 40) % NHEADS;
  const int which = (idx / (40 * NHEADS)) % 2;       // 0=q 1=k
  const int n     = idx / (40 * NHEADS * 2);
  float* base = qkv + (size_t)n * (3 * DMODEL) + which * DMODEL + h * HDIM;
  const float x1 = base[d], x2 = base[d + 40];
  const float c1 = cosp[n * HDIM + d],      s1 = sinp[n * HDIM + d];
  const float c2 = cosp[n * HDIM + d + 40], s2 = sinp[n * HDIM + d + 40];
  base[d]      = x1 * c1 - x2 * s1;
  base[d + 40] = x2 * c2 + x1 * s2;
}

// ---------------------------------------------------------------------------
// Flash attention. grid = (NTOK/64, NHEADS), block = 128 (4 waves).
// Each wave owns 16 queries; 32-key chunks staged in shared LDS.
// Head dim 80 zero-padded to 96 (3 f16 K-chunks of 32).
// Q and P tiles stored swizzled (A-type); K swizzled too; V transposed.
// ---------------------------------------------------------------------------
__global__ __launch_bounds__(128) void attn_kernel(
    const float* __restrict__ qkv, float* __restrict__ outp)
{
  __shared__ _Float16 Qs[64][96];        // swizzled per 32-chunk
  __shared__ _Float16 Ks[32][96];        // swizzled per 32-chunk
  __shared__ _Float16 Vt[HDIM][32];      // transposed: [dim][key]
  __shared__ _Float16 Ps[4][16][32];     // swizzled

  const int tid = threadIdx.x;
  const int lane = tid & 31, wave = tid >> 5;
  const int laneLo = lane & 15, laneHi = lane >> 4;
  const int head  = blockIdx.y;
  const int qbase = blockIdx.x * 64;
  const float scale = 0.11180339887498949f;   // 80^-0.5
  const h4 zero4 = {};

  // stage Q tile (pre-scaled, swizzled): batched loads then stores
  {
    float4 vq[10];
#pragma unroll
    for (int it = 0; it < 10; ++it) {
      const int i = tid + it * 128;
      const int r = i / 20, c4 = i % 20;
      vq[it] = *(const float4*)(qkv + (size_t)(qbase + r) * (3 * DMODEL)
                                + head * HDIM + c4 * 4);
    }
#pragma unroll
    for (int it = 0; it < 10; ++it) {
      const int i = tid + it * 128;
      const int r = i / 20, c4 = i % 20;
      *(h4*)&Qs[r][swz(c4 * 4)] = cvt4(vq[it], scale);
    }
#pragma unroll
    for (int it = 0; it < 2; ++it) {
      const int i = tid + it * 128;
      const int r = i >> 2, c = i & 3;
      *(h4*)&Qs[r][swz(80 + c * 4)] = zero4;
    }
  }
  __syncthreads();

  // per-wave Q fragments (3 K-chunks), rows wave*16..+15
  v16h qf[3];
  {
    const int row = wave * 16 + laneLo;
#pragma unroll
    for (int c = 0; c < 3; ++c)
      qf[c] = load_frag32(&Qs[row][c * 32 + laneHi * 16]);
  }

  v8f oacc[5] = {};
  float mrow[8], lrow[8];
#pragma unroll
  for (int j = 0; j < 8; ++j) { mrow[j] = -1e30f; lrow[j] = 0.0f; }

  // swizzled P columns for this lane (see swz): p0 at col laneLo, p1 at +16
  const int pcol0 = (laneLo & 7) + ((laneLo & 8) << 1);
  const int pcol1 = pcol0 + 8;

  for (int kt = 0; kt < NTOK; kt += 32) {
    __syncthreads();
    // stage K (swizzled) and V (transposed): batched loads then stores
    {
      float4 vk[5], vv[5];
#pragma unroll
      for (int it = 0; it < 5; ++it) {
        const int i = tid + it * 128;
        const int r = i / 20, c4 = i % 20;
        vk[it] = *(const float4*)(qkv + (size_t)(kt + r) * (3 * DMODEL)
                                  + DMODEL + head * HDIM + c4 * 4);
        vv[it] = *(const float4*)(qkv + (size_t)(kt + r) * (3 * DMODEL)
                                  + 2 * DMODEL + head * HDIM + c4 * 4);
      }
#pragma unroll
      for (int it = 0; it < 5; ++it) {
        const int i = tid + it * 128;
        const int r = i / 20, c4 = i % 20;
        *(h4*)&Ks[r][swz(c4 * 4)] = cvt4(vk[it], 1.0f);
        const h4 hv = cvt4(vv[it], 1.0f);
        Vt[c4 * 4 + 0][swz(r)] = hv[0];
        Vt[c4 * 4 + 1][swz(r)] = hv[1];
        Vt[c4 * 4 + 2][swz(r)] = hv[2];
        Vt[c4 * 4 + 3][swz(r)] = hv[3];
      }
      {
        const int r = tid >> 2, c = tid & 3;   // K pad: 32 rows x 4 h4
        *(h4*)&Ks[r][swz(80 + c * 4)] = zero4;
      }
    }
    __syncthreads();

    // scores: two 16x16 tiles (keys 0-15, 16-31)
    v8f s0 = {}, s1 = {};
    const int koff = laneHi * 16;
#pragma unroll
    for (int c = 0; c < 3; ++c) {
      const v16h b0 = load_frag32(&Ks[laneLo][c * 32 + koff]);
      const v16h b1 = load_frag32(&Ks[16 + laneLo][c * 32 + koff]);
      s0 = WMMA_F16(qf[c], b0, s0);
      s1 = WMMA_F16(qf[c], b1, s1);
    }

    // online softmax (rows live across VGPR index j; 16-lane butterflies)
#pragma unroll
    for (int j = 0; j < 8; ++j) {
      float v0 = s0[j], v1 = s1[j];
      float mx = fmaxf(v0, v1);
#pragma unroll
      for (int off = 1; off < 16; off <<= 1) mx = fmaxf(mx, __shfl_xor(mx, off, 32));
      const float mnew  = fmaxf(mrow[j], mx);
      const float alpha = __expf(mrow[j] - mnew);
      const float p0 = __expf(v0 - mnew), p1 = __expf(v1 - mnew);
      float rs = p0 + p1;
#pragma unroll
      for (int off = 1; off < 16; off <<= 1) rs += __shfl_xor(rs, off, 32);
      lrow[j] = lrow[j] * alpha + rs;
      mrow[j] = mnew;
#pragma unroll
      for (int t = 0; t < 5; ++t) oacc[t][j] *= alpha;
      Ps[wave][j + laneHi * 8][pcol0] = (_Float16)p0;
      Ps[wave][j + laneHi * 8][pcol1] = (_Float16)p1;
    }

    // P fragment (swizzled A layout): 32 contiguous bytes
    const v16h pf = load_frag32(&Ps[wave][laneLo][laneHi * 16]);
    // O += P @ V  (5 tiles of 16 dims), V fragments contiguous in Vt
#pragma unroll
    for (int t = 0; t < 5; ++t) {
      const v16h vb = load_frag32(&Vt[t * 16 + laneLo][koff]);
      oacc[t] = WMMA_F16(pf, vb, oacc[t]);
    }
  }

  // normalize and write out[q][head*80 + dim]
#pragma unroll
  for (int t = 0; t < 5; ++t) {
    const int dim = t * 16 + laneLo;
#pragma unroll
    for (int j = 0; j < 8; ++j) {
      const int q = qbase + wave * 16 + j + laneHi * 8;
      outp[(size_t)q * DMODEL + head * HDIM + dim] = oacc[t][j] / lrow[j];
    }
  }
}

// ---------------------------------------------------------------------------
// act = silu(g) * u, written into g (vectorized; total % 4 == 0)
// ---------------------------------------------------------------------------
__global__ void silu_mul_kernel(float* __restrict__ g,
                                const float* __restrict__ u, int total4)
{
  const int i = blockIdx.x * blockDim.x + threadIdx.x;
  if (i >= total4) return;
  float4 gv = *(const float4*)(g + (size_t)i * 4);
  const float4 uv = *(const float4*)(u + (size_t)i * 4);
  gv.x = (gv.x / (1.0f + __expf(-gv.x))) * uv.x;
  gv.y = (gv.y / (1.0f + __expf(-gv.y))) * uv.y;
  gv.z = (gv.z / (1.0f + __expf(-gv.z))) * uv.z;
  gv.w = (gv.w / (1.0f + __expf(-gv.w))) * uv.w;
  *(float4*)(g + (size_t)i * 4) = gv;
}

// ---------------------------------------------------------------------------
// Host-side orchestration
// ---------------------------------------------------------------------------
extern "C" void kernel_launch(void* const* d_in, const int* in_sizes, int n_in,
                              void* d_out, int out_size, void* d_ws, size_t ws_size,
                              hipStream_t stream)
{
  (void)in_sizes; (void)n_in; (void)out_size; (void)ws_size;
  const float* pixel   = (const float*)d_in[0];
  const float* cosp    = (const float*)d_in[1];
  const float* sinp    = (const float*)d_in[2];
  const float* patch_w = (const float*)d_in[3];
  const float* norm1_w = (const float*)d_in[4];
  const float* qkv_w   = (const float*)d_in[5];
  const float* qkv_b   = (const float*)d_in[6];
  const float* proj_w  = (const float*)d_in[7];
  const float* proj_b  = (const float*)d_in[8];
  const float* norm2_w = (const float*)d_in[9];
  const float* gate_w  = (const float*)d_in[10];
  const float* gate_b  = (const float*)d_in[11];
  const float* up_w    = (const float*)d_in[12];
  const float* up_b    = (const float*)d_in[13];
  const float* down_w  = (const float*)d_in[14];
  const float* down_b  = (const float*)d_in[15];
  const float* lnq_w   = (const float*)d_in[16];
  const float* m1_w    = (const float*)d_in[17];
  const float* m1_b    = (const float*)d_in[18];
  const float* m2_w    = (const float*)d_in[19];
  const float* m2_b    = (const float*)d_in[20];
  float* out = (float*)d_out;

  float* ws   = (float*)d_ws;
  float* x    = ws;                                   // 1600*1280
  float* h    = x    + (size_t)NTOK * DMODEL;         // 1600*1280
  float* qkv  = h    + (size_t)NTOK * DMODEL;         // 1600*3840
  float* ob   = qkv  + (size_t)NTOK * 3 * DMODEL;     // 1600*1280
  float* gbuf = ob   + (size_t)NTOK * DMODEL;         // 1600*3420
  float* ubuf = gbuf + (size_t)NTOK * INTER;          // 1600*3420
  float* h2   = ubuf + (size_t)NTOK * INTER;          // 400*5120

  auto gemm = [&](float* C, const float* A, const float* W, const float* bias,
                  const float* res, int M, int N, int K, int act) {
    dim3 grid((M + 63) / 64, (N + 127) / 128);
    gemm_f16_kernel<<<grid, 128, 0, stream>>>(A, W, bias, res, C, M, N, K, act);
  };

  // patch embed: x = pixel @ patch_w^T
  gemm(x, pixel, patch_w, nullptr, nullptr, NTOK, DMODEL, PATCH_IN, 0);

  for (int i = 0; i < DEPTH; ++i) {
    const float* q_w = qkv_w  + (size_t)i * 3 * DMODEL * DMODEL;
    const float* q_b = qkv_b  + (size_t)i * 3 * DMODEL;
    const float* p_w = proj_w + (size_t)i * DMODEL * DMODEL;
    const float* p_b = proj_b + (size_t)i * DMODEL;
    const float* g_w = gate_w + (size_t)i * INTER * DMODEL;
    const float* g_b = gate_b + (size_t)i * INTER;
    const float* u_w = up_w   + (size_t)i * INTER * DMODEL;
    const float* u_b = up_b   + (size_t)i * INTER;
    const float* d_w = down_w + (size_t)i * DMODEL * INTER;
    const float* d_b = down_b + (size_t)i * DMODEL;

    rmsnorm_kernel<<<NTOK, 256, 0, stream>>>(x, norm1_w + (size_t)i * DMODEL, h, DMODEL);
    gemm(qkv, h, q_w, q_b, nullptr, NTOK, 3 * DMODEL, DMODEL, 0);
    {
      const int total = NTOK * 2 * NHEADS * 40;
      rope_kernel<<<(total + 255) / 256, 256, 0, stream>>>(qkv, cosp, sinp);
    }
    attn_kernel<<<dim3(NTOK / 64, NHEADS), 128, 0, stream>>>(qkv, ob);
    gemm(x, ob, p_w, p_b, x, NTOK, DMODEL, DMODEL, 0);        // residual add

    rmsnorm_kernel<<<NTOK, 256, 0, stream>>>(x, norm2_w + (size_t)i * DMODEL, h, DMODEL);
    gemm(gbuf, h, g_w, g_b, nullptr, NTOK, INTER, DMODEL, 0);
    gemm(ubuf, h, u_w, u_b, nullptr, NTOK, INTER, DMODEL, 0);
    {
      const int total4 = (NTOK * INTER) / 4;
      silu_mul_kernel<<<(total4 + 255) / 256, 256, 0, stream>>>(gbuf, ubuf, total4);
    }
    gemm(x, gbuf, d_w, d_b, x, NTOK, DMODEL, INTER, 0);       // residual add
  }

  // merger: rms -> reshape (400,5120) -> gelu(m1) -> m2
  rmsnorm_kernel<<<NTOK, 256, 0, stream>>>(x, lnq_w, h, DMODEL);
  gemm(h2, h, m1_w, m1_b, nullptr, MROWS, MDIM, MDIM, /*gelu=*/1);
  gemm(out, h2, m2_w, m2_b, nullptr, MROWS, OUTD, MDIM, 0);
}
